// OriLinearGNN_47201690583805
// MI455X (gfx1250) — matrix operand order repaired and checked by reading, not compiled, not run on hardware
//
#include <hip/hip_runtime.h>
#include <hip/hip_bf16.h>

// ---------------------------------------------------------------------------
// Sizes from the reference
// ---------------------------------------------------------------------------
constexpr int V  = 50000;
constexpr int E  = 150000;
constexpr int LN = 256;
constexpr int S  = 32;
constexpr int T  = 8;
constexpr int NK = 1024;      // S*S outputs of the Xi transform
constexpr int KK = 512;       // 2*LN (one-hot part handled as gather-add)
constexpr float MU_OVER_S = 0.9f / 32.0f;

// trivially-constructible vector types (safe inside unions, unlike HIP's uint4)
typedef __attribute__((ext_vector_type(16))) _Float16 v16h;
typedef __attribute__((ext_vector_type(8)))  float    v8f;
typedef __attribute__((ext_vector_type(4)))  unsigned int u32x4;
typedef __attribute__((ext_vector_type(4)))  float    f32x4;

union FragH { u32x4 u[2]; v16h h; };
union ARow  { u32x4 u[4]; _Float16 f[32]; };

// async global->LDS copy of 16B: one base pair + compile-time offset
// (INST_OFFSET is added to BOTH the LDS and the global address)
#define ASYNC_LDS_B128(ldsoff, gptr, imm)                                     \
    asm volatile("global_load_async_to_lds_b128 %0, %1, off offset:%2"        \
                 :: "v"(ldsoff), "v"(gptr), "n"(imm) : "memory")

// ---------------------------------------------------------------------------
// Workspace layout (bytes)
// ---------------------------------------------------------------------------
constexpr size_t OFF_FEATH = 0;                                     // V*LN f16
constexpr size_t OFF_WHX   = OFF_FEATH + (size_t)V * LN * 2;        // 1024x512 f16
constexpr size_t OFF_AH    = OFF_WHX   + (size_t)NK * KK * 2;       // E*1024 f16
constexpr size_t OFF_BEDGE = OFF_AH    + (size_t)E * NK * 2;        // E*32 f32
constexpr size_t OFF_H0    = OFF_BEDGE + (size_t)E * S * 4;         // V*32 f32
constexpr size_t OFF_H1    = OFF_H0    + (size_t)V * S * 4;         // V*32 f32
constexpr size_t OFF_LOG   = OFF_H1    + (size_t)V * S * 4;         // V f32

// ---------------------------------------------------------------------------
// Small prep kernels
// ---------------------------------------------------------------------------
__global__ void f2h_kernel(const float* __restrict__ in, _Float16* __restrict__ out, int n) {
    int i = blockIdx.x * 256 + threadIdx.x;
    if (i < n) out[i] = (_Float16)in[i];
}

// pack W_xi[:, :512] (row stride 520) into dense f16 [1024][512]
__global__ void packw_kernel(const float* __restrict__ W_xi, _Float16* __restrict__ Whx) {
    int i = blockIdx.x * 256 + threadIdx.x;           // 1024*512 total
    int n = i >> 9, k = i & 511;
    Whx[i] = (_Float16)W_xi[(size_t)n * 520 + k];
}

__device__ __forceinline__ float hash_u01(unsigned int x) {
    x ^= x >> 16; x *= 0x7feb352du;
    x ^= x >> 15; x *= 0x846ca68bu;
    x ^= x >> 16;
    return (float)(x >> 8) * (1.0f / 16777216.0f);
}

__global__ void hinit_kernel(float* __restrict__ H, int n) {
    int i = blockIdx.x * 256 + threadIdx.x;
    if (i < n) H[i] = hash_u01((unsigned int)i * 2654435761u + 0x9e3779b9u);
}

__global__ void zero_kernel(float* __restrict__ p, int n) {
    int i = blockIdx.x * 256 + threadIdx.x;
    if (i < n) p[i] = 0.0f;
}

// ---------------------------------------------------------------------------
// Xi GEMM: A_h[e, n] = tanh( Xcat[e,:512] @ Whx[n,:]^T + b_xi[n]
//                            + W_xi[n, 512 + et[e]] ) * (mu/S)/dg[e]
//
// block = 32 edges (2 M-tiles), 512 threads (16 waves).
// Wave w: M-tile (w>>3), N-tiles (w&7)*8 .. +7  -> 8 accumulators (64 VGPRs).
// X tile (32x512 f16) gathered once via async global->LDS.
// B panel (all 1024 n x 32 k of the current k-step, 64 KB) staged in LDS,
// double-buffered: panel kb+1 prefetched with async global->LDS while WMMAs
// consume panel kb via ds_load_b128. One s_wait_asynccnt + barrier per k-step.
// ---------------------------------------------------------------------------
__global__ __launch_bounds__(512, 1) void gemm_xi_kernel(
    const _Float16* __restrict__ feat_h,   // [V,256] f16
    const _Float16* __restrict__ Whx,      // [1024,512] f16
    const float*    __restrict__ W_xi,     // [1024,520] f32 (for one-hot cols)
    const float*    __restrict__ b_xi,     // [1024]
    const int*      __restrict__ X_Node,   // 1-based
    const int*      __restrict__ X_Neis,   // 1-based
    const int*      __restrict__ etype,    // 1-based
    const float*    __restrict__ dg,
    _Float16*       __restrict__ Ah)       // [E,1024] f16
{
    __shared__ __align__(16) _Float16 Xtile[32][KK];    // 32 KB
    __shared__ __align__(16) _Float16 Bp[2][NK][32];    // 2 x 64 KB
    __shared__ float scS[32];
    __shared__ int   etS[32];

    const int tid   = threadIdx.x;
    const int ebase = blockIdx.x * 32;

    if (tid < 32) {
        int e = ebase + tid;
        int eg = e < E ? e : E - 1;
        scS[tid] = MU_OVER_S / dg[eg];
        etS[tid] = etype[eg] - 1;
    }
    // cooperative gather: row r = concat(feat[src], feat[nei]) in f16,
    // via async global->LDS (per-lane global addr -> per-lane LDS offset)
    {
        int r    = tid >> 4;            // 0..31
        int cseg = (tid & 15) * 32;     // 0..480
        int e    = ebase + r;
        int eg   = e < E ? e : E - 1;
        int node = (cseg < LN ? X_Node[eg] : X_Neis[eg]) - 1;
        int col  = (cseg < LN ? cseg : cseg - LN);
        const _Float16* gsrc = feat_h + (size_t)node * LN + col;
        unsigned lds0 = (unsigned)(size_t)(void*)&Xtile[r][cseg];
        ASYNC_LDS_B128(lds0, gsrc, 0);
        ASYNC_LDS_B128(lds0, gsrc, 16);
        ASYNC_LDS_B128(lds0, gsrc, 32);
        ASYNC_LDS_B128(lds0, gsrc, 48);
    }
    // stage B panel for kb = 0 (each thread: 2 rows x 64 B)
    {
        #pragma unroll
        for (int rr = 0; rr < 2; ++rr) {
            int n = tid * 2 + rr;
            const _Float16* gsrc = Whx + (size_t)n * KK;      // kb = 0
            unsigned lds0 = (unsigned)(size_t)(void*)&Bp[0][n][0];
            ASYNC_LDS_B128(lds0, gsrc, 0);
            ASYNC_LDS_B128(lds0, gsrc, 16);
            ASYNC_LDS_B128(lds0, gsrc, 32);
            ASYNC_LDS_B128(lds0, gsrc, 48);
        }
    }
    asm volatile("s_wait_asynccnt 0" ::: "memory");
    __syncthreads();

    const int wave   = tid >> 5;          // 0..15
    const int lane   = tid & 31;
    const int half   = lane >> 4;
    const int r16    = lane & 15;
    const int mrow0  = (wave >> 3) * 16;  // 0 or 16: M-tile of this wave
    const int ntbase = (wave & 7) * 8;    // 8 N-tiles of this wave

    v8f acc[8];
    #pragma unroll
    for (int nt = 0; nt < 8; ++nt) acc[nt] = (v8f){0,0,0,0,0,0,0,0};

    #pragma unroll 1
    for (int kb = 0; kb < KK / 32; ++kb) {
        const int cur = kb & 1;
        // prefetch next B panel while computing on the current one
        if (kb < KK / 32 - 1) {
            const int nxt = cur ^ 1;
            #pragma unroll
            for (int rr = 0; rr < 2; ++rr) {
                int n = tid * 2 + rr;
                const _Float16* gsrc = Whx + (size_t)n * KK + (kb + 1) * 32;
                unsigned lds0 = (unsigned)(size_t)(void*)&Bp[nxt][n][0];
                ASYNC_LDS_B128(lds0, gsrc, 0);
                ASYNC_LDS_B128(lds0, gsrc, 16);
                ASYNC_LDS_B128(lds0, gsrc, 32);
                ASYNC_LDS_B128(lds0, gsrc, 48);
            }
        }
        // A fragment (16x32 f16): lane<16 K {0..7,16..23}, lane>=16 K {8..15,24..31}
        FragH a;
        a.u[0] = *(const u32x4*)(&Xtile[mrow0 + r16][kb * 32 +      8 * half]);
        a.u[1] = *(const u32x4*)(&Xtile[mrow0 + r16][kb * 32 + 16 + 8 * half]);
        #pragma unroll
        for (int nt = 0; nt < 8; ++nt) {
            int n = (ntbase + nt) * 16 + r16;
            // B fragment from LDS: lane<16 -> col n, K {0..7,16..23}; lane>=16 -> K {8..15,24..31}
            FragH b;
            const _Float16* wr = &Bp[cur][n][8 * half];
            b.u[0] = *(const u32x4*)(wr);
            b.u[1] = *(const u32x4*)(wr + 16);
            acc[nt] = __builtin_amdgcn_wmma_f32_16x16x32_f16(
                false, a.h, false, b.h, (short)0, acc[nt], false, false);
        }
        // prefetch (issued a full k-step ago) must land before next iteration
        asm volatile("s_wait_asynccnt 0" ::: "memory");
        __syncthreads();
    }

    // epilogue: + b_xi + one-hot column of W_xi, tanh, per-edge scale, store
    // f16. Fully unrolled: constant indices keep acc[] in registers.
    #pragma unroll
    for (int nt = 0; nt < 8; ++nt) {
        int n = (ntbase + nt) * 16 + r16;
        float bias = b_xi[n];
        const float* wet = W_xi + (size_t)n * 520 + 512;
        #pragma unroll
        for (int i = 0; i < 8; ++i) {
            int mrow = mrow0 + i + 8 * half;            // edge row in tile
            int e = ebase + mrow;
            float v = acc[nt][i] + bias + wet[etS[mrow]];
            v = tanhf(v) * scS[mrow];
            if (e < E) Ah[(size_t)e * NK + n] = (_Float16)v;
        }
    }
}

// ---------------------------------------------------------------------------
// b_edge[e,s] = tanh( feat[nei[e]] . W_rou[s] + b_rou[s] )
// block = 8 edges x 32 s
// ---------------------------------------------------------------------------
__global__ __launch_bounds__(256) void rou_kernel(
    const float* __restrict__ feat, const float* __restrict__ W_rou,
    const float* __restrict__ b_rou, const int* __restrict__ X_Neis,
    float* __restrict__ bedge)
{
    __shared__ __align__(16) float fs[8][LN];   // 8 KB
    int tid = threadIdx.x;
    int ebase = blockIdx.x * 8;
    {
        int r  = tid >> 5;            // 0..7
        int c0 = (tid & 31) * 8;      // 0..248
        int node = X_Neis[ebase + r] - 1;
        const f32x4* s = (const f32x4*)(feat + (size_t)node * LN + c0);
        f32x4* d = (f32x4*)(&fs[r][c0]);
        d[0] = s[0]; d[1] = s[1];
    }
    __syncthreads();
    int el = tid >> 5;
    int s  = tid & 31;
    const float* wr = W_rou + (size_t)s * LN;
    float sum = b_rou[s];
    #pragma unroll 4
    for (int k = 0; k < LN; ++k) sum += fs[el][k] * wr[k];
    bedge[(size_t)(ebase + el) * S + s] = tanhf(sum);
}

// ---------------------------------------------------------------------------
// one propagation step: He = A[e] @ H[src[e]] + b[e]; scatter-add at raw
// 1-based X_Neis (index V dropped, node 0 untouched) == segment_sum semantics
// one wave per edge, lane = output row
// ---------------------------------------------------------------------------
__global__ __launch_bounds__(256) void prop_kernel(
    const _Float16* __restrict__ Ah, const float* __restrict__ bedge,
    const int* __restrict__ X_Node, const int* __restrict__ X_Neis,
    const float* __restrict__ Hin, float* __restrict__ Hout)
{
    int tid  = threadIdx.x;
    int lane = tid & 31;
    int e    = blockIdx.x * 8 + (tid >> 5);
    int src  = X_Node[e] - 1;
    float h  = Hin[(size_t)src * S + lane];

    ARow arow;
    const u32x4* ap = (const u32x4*)(Ah + (size_t)e * NK + lane * S);
    #pragma unroll
    for (int q = 0; q < 4; ++q) arow.u[q] = ap[q];

    float sum = bedge[(size_t)e * S + lane];
    #pragma unroll
    for (int j = 0; j < S; ++j)
        sum += (float)arow.f[j] * __shfl(h, j, 32);

    int idx = X_Neis[e];                  // raw 1-based; idx==V dropped
    if (idx < V) atomicAdd(&Hout[(size_t)idx * S + lane], sum);
}

// ---------------------------------------------------------------------------
// finale: logits = H @ W_l1 + b; softmax over V; out = tanh(sum H*attn)
// single 1024-thread workgroup, LDS reductions
// ---------------------------------------------------------------------------
__global__ __launch_bounds__(1024) void final_kernel(
    const float* __restrict__ H, const float* __restrict__ W_l1,
    const float* __restrict__ b_l1, float* __restrict__ logits,
    float* __restrict__ out)
{
    __shared__ float red[1024];
    __shared__ float accS[S];
    int tid = threadIdx.x;

    float w[S];
    #pragma unroll
    for (int s = 0; s < S; ++s) w[s] = W_l1[s];
    float bl = b_l1[0];

    float lmax = -1e30f;
    for (int v = tid; v < V; v += 1024) {
        const float* hv = H + (size_t)v * S;
        float dot = bl;
        #pragma unroll
        for (int s = 0; s < S; ++s) dot += hv[s] * w[s];
        logits[v] = dot;
        lmax = fmaxf(lmax, dot);
    }
    red[tid] = lmax; __syncthreads();
    for (int st = 512; st > 0; st >>= 1) {
        if (tid < st) red[tid] = fmaxf(red[tid], red[tid + st]);
        __syncthreads();
    }
    float gmax = red[0];
    __syncthreads();

    if (tid < S) accS[tid] = 0.0f;
    __syncthreads();

    float lsum = 0.0f;
    float acc[S];
    #pragma unroll
    for (int s = 0; s < S; ++s) acc[s] = 0.0f;
    for (int v = tid; v < V; v += 1024) {
        float ez = __expf(logits[v] - gmax);
        lsum += ez;
        const float* hv = H + (size_t)v * S;
        #pragma unroll
        for (int s = 0; s < S; ++s) acc[s] += ez * hv[s];
    }
    red[tid] = lsum; __syncthreads();
    for (int st = 512; st > 0; st >>= 1) {
        if (tid < st) red[tid] += red[tid + st];
        __syncthreads();
    }
    float gsum = red[0];
    #pragma unroll
    for (int s = 0; s < S; ++s) atomicAdd(&accS[s], acc[s]);
    __syncthreads();
    if (tid < S) out[tid] = tanhf(accS[tid] / gsum);
}

// ---------------------------------------------------------------------------
// driver
// ---------------------------------------------------------------------------
extern "C" void kernel_launch(void* const* d_in, const int* in_sizes, int n_in,
                              void* d_out, int out_size, void* d_ws, size_t ws_size,
                              hipStream_t stream)
{
    const float* feat   = (const float*)d_in[0];
    const float* W_xi   = (const float*)d_in[1];
    const float* b_xi   = (const float*)d_in[2];
    const float* W_rou  = (const float*)d_in[3];
    const float* b_rou  = (const float*)d_in[4];
    const float* W_l1   = (const float*)d_in[5];
    const float* b_l1   = (const float*)d_in[6];
    const int*   X_Node = (const int*)d_in[7];
    const int*   X_Neis = (const int*)d_in[8];
    const int*   etype  = (const int*)d_in[9];
    const float* dg     = (const float*)d_in[10];
    float* out = (float*)d_out;

    char* ws = (char*)d_ws;
    _Float16* feat_h = (_Float16*)(ws + OFF_FEATH);
    _Float16* Whx    = (_Float16*)(ws + OFF_WHX);
    _Float16* Ah     = (_Float16*)(ws + OFF_AH);
    float*    bedge  = (float*)(ws + OFF_BEDGE);
    float*    H0     = (float*)(ws + OFF_H0);
    float*    H1     = (float*)(ws + OFF_H1);
    float*    logits = (float*)(ws + OFF_LOG);

    f2h_kernel<<<(V * LN + 255) / 256, 256, 0, stream>>>(feat, feat_h, V * LN);
    packw_kernel<<<(NK * KK) / 256, 256, 0, stream>>>(W_xi, Whx);
    hinit_kernel<<<(V * S + 255) / 256, 256, 0, stream>>>(H0, V * S);

    gemm_xi_kernel<<<(E + 31) / 32, 512, 0, stream>>>(feat_h, Whx, W_xi, b_xi,
                                                      X_Node, X_Neis, etype, dg, Ah);
    rou_kernel<<<E / 8, 256, 0, stream>>>(feat, W_rou, b_rou, X_Neis, bedge);

    float* Hin = H0; float* Hout = H1;
    for (int t = 0; t < T; ++t) {
        zero_kernel<<<(V * S) / 256, 256, 0, stream>>>(Hout, V * S);
        prop_kernel<<<E / 8, 256, 0, stream>>>(Ah, bedge, X_Node, X_Neis, Hin, Hout);
        float* tmp = Hin; Hin = Hout; Hout = tmp;
    }

    final_kernel<<<1, 1024, 0, stream>>>(Hin, W_l1, b_l1, logits, out);
}
